// Model_83580063580842
// MI455X (gfx1250) — matrix operand order repaired and checked by reading, not compiled, not run on hardware
//
#include <hip/hip_runtime.h>
#include <math.h>

typedef _Float16 v16h __attribute__((ext_vector_type(16)));
typedef _Float16 v8h  __attribute__((ext_vector_type(8)));
typedef float    v8f  __attribute__((ext_vector_type(8)));

#define HEADS 4
#define DIMC  128          // node feature width for both layers (4*32)
#define NEG_SLOPE 0.2f
#define LN_EPS 1e-5f
#define EPS_SOFTMAX 1e-16f
#define LDS_STRIDE 136     // halves per column: 272B row stride -> 16B aligned,
                           // lane l hits bank 4*l mod 64 => conflict-free b128 reads

// ---- order-preserving float<->uint for atomicMax-based segment_max ----
__device__ __forceinline__ unsigned enc_f32(float f) {
  unsigned u = __float_as_uint(f);
  return (u & 0x80000000u) ? ~u : (u | 0x80000000u);
}
__device__ __forceinline__ float dec_f32(unsigned e) {
  return (e & 0x80000000u) ? __uint_as_float(e & 0x7FFFFFFFu)
                           : __uint_as_float(~e);
}
#define ENC_NEG_INF 0x007FFFFFu   // enc_f32(-inf)

// =====================================================================
// GEMM: C[M x 128] = A[M x 128] @ B[128 x 128], fp32 in/out, f16 WMMA.
// Block = 256 thr = 8 waves; block tile = 64 rows x 128 cols.
// W is staged once per block into LDS (f16, col-major, padded), so each
// B fragment is two conflict-free ds_load_b128.  Wave w computes row
// tile (w>>1) x col tiles [64*(w&1) .. +64): 4 accumulators, A fragment
// shared across them -> 16 v_wmma per wave per kernel.
// =====================================================================
__global__ void __launch_bounds__(256)
gemm128_wmma(const float* __restrict__ A, const float* __restrict__ B,
             float* __restrict__ C, int M)
{
  __shared__ _Float16 lb[128 * LDS_STRIDE];   // lb[c*LDS_STRIDE + k]

  // ---- stage W -> LDS, f16 col-major (coalesced global reads) ----
  for (int i = threadIdx.x; i < 128 * 128; i += 256) {
    const int r = i >> 7, c = i & 127;
    lb[c * LDS_STRIDE + r] = (_Float16)B[i];
  }
  __syncthreads();

  const int wave = threadIdx.x >> 5;
  const int lane = threadIdx.x & 31;
  const int rt   = wave >> 1;              // row tile 0..3
  const int ch   = wave & 1;               // col half (4 tiles each)
  const int m0   = blockIdx.x * 64 + rt * 16;
  const int hsel = lane >> 4;              // 0 = lanes 0-15, 1 = lanes 16-31
  const int lr   = lane & 15;
  const int arow = m0 + lr;
  const bool aok = arow < M;
  const float* __restrict__ ap = A + (size_t)arow * DIMC;

  v8f acc0 = {}, acc1 = {}, acc2 = {}, acc3 = {};
  #pragma unroll
  for (int k0 = 0; k0 < DIMC; k0 += 32) {
    // A fragment (16x32 f16): lane<16 holds K=[k0..k0+8) & [k0+16..k0+24),
    // lane>=16 holds K=[k0+8..k0+16) & [k0+24..k0+32)  (ISA 05_wmma layout)
    v16h af = {};
    if (aok) {
      const float4 f0 = *(const float4*)(ap + k0 + 8 * hsel);
      const float4 f1 = *(const float4*)(ap + k0 + 8 * hsel + 4);
      const float4 g0 = *(const float4*)(ap + k0 + 16 + 8 * hsel);
      const float4 g1 = *(const float4*)(ap + k0 + 16 + 8 * hsel + 4);
      af[0]=(_Float16)f0.x; af[1]=(_Float16)f0.y; af[2]=(_Float16)f0.z; af[3]=(_Float16)f0.w;
      af[4]=(_Float16)f1.x; af[5]=(_Float16)f1.y; af[6]=(_Float16)f1.z; af[7]=(_Float16)f1.w;
      af[8]=(_Float16)g0.x; af[9]=(_Float16)g0.y; af[10]=(_Float16)g0.z; af[11]=(_Float16)g0.w;
      af[12]=(_Float16)g1.x; af[13]=(_Float16)g1.y; af[14]=(_Float16)g1.z; af[15]=(_Float16)g1.w;
    }
    // B fragment: lane (hsel,lr) needs B[k0+8*hsel .. +8) & B[k0+16+8*hsel .. +8)[col]
    // in col-major LDS -> two 16B ds loads.
    #pragma unroll
    for (int j = 0; j < 4; ++j) {
      const int col = ch * 64 + j * 16 + lr;
      const int off = col * LDS_STRIDE + k0 + 8 * hsel;
      union { v16h v; v8h h[2]; } bu;
      bu.h[0] = *(const v8h*)&lb[off];
      bu.h[1] = *(const v8h*)&lb[off + 16];
      v8f* accp = (j == 0) ? &acc0 : (j == 1) ? &acc1 : (j == 2) ? &acc2 : &acc3;
      *accp = __builtin_amdgcn_wmma_f32_16x16x32_f16(false, af, false, bu.v,
                                                     (short)0, *accp, false, false);
    }
  }

  // D layout: VGPR v -> row m0 + v + 8*hsel, col tile base + lr
  #pragma unroll
  for (int j = 0; j < 4; ++j) {
    const v8f acc = (j == 0) ? acc0 : (j == 1) ? acc1 : (j == 2) ? acc2 : acc3;
    const int col = ch * 64 + j * 16 + lr;
    #pragma unroll
    for (int v = 0; v < 8; ++v) {
      const int r = m0 + v + 8 * hsel;
      if (r < M) C[(size_t)r * DIMC + col] = acc[v];
    }
  }
}

// ---- c_edge[h] = sum_c We[h*32+c] * a_edge[h*32+c]  (both layers) ----
__global__ void cedge_precompute(const float* __restrict__ We1,
                                 const float* __restrict__ ae1,
                                 const float* __restrict__ We2,
                                 const float* __restrict__ ae2,
                                 float* __restrict__ ced /* 8 floats */)
{
  __shared__ float s[256];
  const int t = threadIdx.x;
  s[t] = (t < 128) ? We1[t] * ae1[t] : We2[t - 128] * ae2[t - 128];
  __syncthreads();
  if ((t & 31) == 0) {
    float sum = 0.f;
    #pragma unroll
    for (int i = 0; i < 32; ++i) sum += s[t + i];
    ced[t >> 5] = sum;   // 0..3 = layer1 heads, 4..7 = layer2 heads
  }
}

// ---- per-(node,head) attention dots ----
__global__ void node_attn(const float* __restrict__ h,
                          const float* __restrict__ att_src,
                          const float* __restrict__ att_dst,
                          float* __restrict__ asrc, float* __restrict__ adst,
                          int n)
{
  const int idx = blockIdx.x * blockDim.x + threadIdx.x;
  if (idx >= n * HEADS) return;
  const int node = idx >> 2, hd = idx & 3;
  const float* __restrict__ hp = h + (size_t)node * DIMC + hd * 32;
  const float* __restrict__ as = att_src + hd * 32;
  const float* __restrict__ ad = att_dst + hd * 32;
  float s = 0.f, d = 0.f;
  #pragma unroll
  for (int c = 0; c < 32; ++c) { const float v = hp[c]; s += v * as[c]; d += v * ad[c]; }
  asrc[idx] = s; adst[idx] = d;
}

// ---- zero accumulators / init segment-max ----
__global__ void init_layer(float* __restrict__ outacc, float* __restrict__ denom,
                           unsigned* __restrict__ amaxenc, int n)
{
  const int i = blockIdx.x * blockDim.x + threadIdx.x;
  if (i < n * DIMC) outacc[i] = 0.f;
  if (i < n * HEADS) { denom[i] = 0.f; amaxenc[i] = ENC_NEG_INF; }
}

// ---- alpha = leaky_relu(asrc[src]+adst[dst]+ew*c[h]); segment_max ----
__global__ void edge_alpha_max(const int* __restrict__ src, const int* __restrict__ dst,
                               const float* __restrict__ ew,
                               const float* __restrict__ asrc, const float* __restrict__ adst,
                               const float* __restrict__ ced,
                               float* __restrict__ alpha, unsigned* __restrict__ amaxenc,
                               int E)
{
  const int e = blockIdx.x * blockDim.x + threadIdx.x;
  if (e >= E) return;
  const int s = src[e], d = dst[e];
  const float w = ew[e];
  #pragma unroll
  for (int hd = 0; hd < HEADS; ++hd) {
    float a = asrc[s * HEADS + hd] + adst[d * HEADS + hd] + w * ced[hd];
    a = (a > 0.f) ? a : NEG_SLOPE * a;
    alpha[(size_t)e * HEADS + hd] = a;
    atomicMax(&amaxenc[d * HEADS + hd], enc_f32(a));
  }
}

// ---- decode max; isfinite -> 0 (reference semantics) ----
__global__ void finalize_amax(const unsigned* __restrict__ amaxenc,
                              float* __restrict__ amax, int n)
{
  const int i = blockIdx.x * blockDim.x + threadIdx.x;
  if (i >= n * HEADS) return;
  const float f = dec_f32(amaxenc[i]);
  amax[i] = isfinite(f) ? f : 0.f;
}

// ---- ex = exp(alpha - amax[dst]); denom[dst] += ex (in-place over alpha) ----
__global__ void edge_exp_sum(const int* __restrict__ dst,
                             float* __restrict__ alpha_ex,
                             const float* __restrict__ amax,
                             float* __restrict__ denom, int E)
{
  const int e = blockIdx.x * blockDim.x + threadIdx.x;
  if (e >= E) return;
  const int d = dst[e];
  #pragma unroll
  for (int hd = 0; hd < HEADS; ++hd) {
    const float ex = expf(alpha_ex[(size_t)e * HEADS + hd] - amax[d * HEADS + hd]);
    alpha_ex[(size_t)e * HEADS + hd] = ex;
    atomicAdd(&denom[d * HEADS + hd], ex);
  }
}

// ---- out[dst] += h[src] * att ; 128 coalesced channels per edge ----
__global__ void __launch_bounds__(256)
edge_message(const int* __restrict__ src, const int* __restrict__ dst,
             const float* __restrict__ h, const float* __restrict__ ex,
             const float* __restrict__ denom, float* __restrict__ outacc, int E)
{
  const int e = blockIdx.x * 2 + (threadIdx.x >> 7);   // 2 edges / 256-thr block
  if (e >= E) return;
  const int c  = threadIdx.x & 127;
  const int hd = c >> 5;
  const int s = src[e], d = dst[e];
  const float att = ex[(size_t)e * HEADS + hd] / (denom[d * HEADS + hd] + EPS_SOFTMAX);
  atomicAdd(outacc + (size_t)d * DIMC + c, h[(size_t)s * DIMC + c] * att);
}

// ---- LayerNorm(+bias1) + ReLU, one 128-thread block per node ----
__global__ void ln_relu(const float* __restrict__ agg, const float* __restrict__ bias,
                        const float* __restrict__ gamma, const float* __restrict__ beta,
                        float* __restrict__ hout)
{
  __shared__ float sred[128];
  const int node = blockIdx.x, c = threadIdx.x;
  const float v = agg[(size_t)node * DIMC + c] + bias[c];
  sred[c] = v; __syncthreads();
  for (int off = 64; off > 0; off >>= 1) { if (c < off) sred[c] += sred[c + off]; __syncthreads(); }
  const float mu = sred[0] * (1.f / 128.f);
  __syncthreads();
  const float dv = v - mu;
  sred[c] = dv * dv; __syncthreads();
  for (int off = 64; off > 0; off >>= 1) { if (c < off) sred[c] += sred[c + off]; __syncthreads(); }
  const float var = sred[0] * (1.f / 128.f);
  const float y = dv * rsqrtf(var + LN_EPS) * gamma[c] + beta[c];
  hout[(size_t)node * DIMC + c] = fmaxf(y, 0.f);
}

// ---- head-mean + bias2 -> d_out [N,32] ----
__global__ void head_mean(const float* __restrict__ agg, const float* __restrict__ bias2,
                          float* __restrict__ out, int n)
{
  const int idx = blockIdx.x * blockDim.x + threadIdx.x;
  if (idx >= n * 32) return;
  const int node = idx >> 5, c = idx & 31;
  const float* __restrict__ p = agg + (size_t)node * DIMC + c;
  out[idx] = 0.25f * (p[0] + p[32] + p[64] + p[96]) + bias2[c];
}

// =====================================================================
extern "C" void kernel_launch(void* const* d_in, const int* in_sizes, int n_in,
                              void* d_out, int out_size, void* d_ws, size_t ws_size,
                              hipStream_t stream)
{
  (void)n_in; (void)out_size; (void)ws_size;
  const float* x   = (const float*)d_in[0];
  const int*   ei  = (const int*)  d_in[1];
  const float* ew  = (const float*)d_in[2];
  const float* W1  = (const float*)d_in[3];
  const float* We1 = (const float*)d_in[4];
  const float* as1 = (const float*)d_in[5];
  const float* ad1 = (const float*)d_in[6];
  const float* ae1 = (const float*)d_in[7];
  const float* b1  = (const float*)d_in[8];
  const float* lng = (const float*)d_in[9];
  const float* lnb = (const float*)d_in[10];
  const float* W2  = (const float*)d_in[11];
  const float* We2 = (const float*)d_in[12];
  const float* as2 = (const float*)d_in[13];
  const float* ad2 = (const float*)d_in[14];
  const float* ae2 = (const float*)d_in[15];
  const float* b2  = (const float*)d_in[16];
  float* out = (float*)d_out;

  const int n = in_sizes[0] / DIMC;   // 50000
  const int e = in_sizes[1] / 2;      // 1600000
  const int* src = ei;
  const int* dst = ei + e;

  // workspace carve-up (floats)
  float* ws = (float*)d_ws;
  size_t o = 0;
  float* bufH  = ws + o; o += (size_t)n * DIMC;    // node features / agg
  float* bufO  = ws + o; o += (size_t)n * DIMC;    // agg / node features
  float* exbuf = ws + o; o += (size_t)e * HEADS;   // alpha then ex
  float* asrc  = ws + o; o += (size_t)n * HEADS;
  float* adst  = ws + o; o += (size_t)n * HEADS;
  float* amax  = ws + o; o += (size_t)n * HEADS;
  float* denom = ws + o; o += (size_t)n * HEADS;
  unsigned* amaxenc = (unsigned*)(ws + o); o += (size_t)n * HEADS;
  float* ced  = ws + o; o += 8;

  const int gN4   = (n * HEADS + 255) / 256;
  const int gN128 = (n * DIMC + 255) / 256;
  const int gE    = (e + 255) / 256;
  const int gEmsg = (e + 1) / 2;
  const int gGemm = (n + 63) / 64;

  cedge_precompute<<<1, 256, 0, stream>>>(We1, ae1, We2, ae2, ced);

  // ---------------- layer 1 (concat) ----------------
  gemm128_wmma<<<gGemm, 256, 0, stream>>>(x, W1, bufH, n);
  node_attn<<<gN4, 256, 0, stream>>>(bufH, as1, ad1, asrc, adst, n);
  init_layer<<<gN128, 256, 0, stream>>>(bufO, denom, amaxenc, n);
  edge_alpha_max<<<gE, 256, 0, stream>>>(src, dst, ew, asrc, adst, ced + 0, exbuf, amaxenc, e);
  finalize_amax<<<gN4, 256, 0, stream>>>(amaxenc, amax, n);
  edge_exp_sum<<<gE, 256, 0, stream>>>(dst, exbuf, amax, denom, e);
  edge_message<<<gEmsg, 256, 0, stream>>>(src, dst, bufH, exbuf, denom, bufO, e);
  ln_relu<<<n, 128, 0, stream>>>(bufO, b1, lng, lnb, bufH);

  // ---------------- layer 2 (head mean) ----------------
  gemm128_wmma<<<gGemm, 256, 0, stream>>>(bufH, W2, bufO, n);     // h2 -> bufO
  node_attn<<<gN4, 256, 0, stream>>>(bufO, as2, ad2, asrc, adst, n);
  init_layer<<<gN128, 256, 0, stream>>>(bufH, denom, amaxenc, n); // bufH = agg2
  edge_alpha_max<<<gE, 256, 0, stream>>>(src, dst, ew, asrc, adst, ced + 4, exbuf, amaxenc, e);
  finalize_amax<<<gN4, 256, 0, stream>>>(amaxenc, amax, n);
  edge_exp_sum<<<gE, 256, 0, stream>>>(dst, exbuf, amax, denom, e);
  edge_message<<<gEmsg, 256, 0, stream>>>(src, dst, bufO, exbuf, denom, bufH, e);
  head_mean<<<(n * 32 + 255) / 256, 256, 0, stream>>>(bufH, b2, out, n);
}
